// CausalSelfAttention_9208409883154
// MI455X (gfx1250) — matrix operand order repaired and checked by reading, not compiled, not run on hardware
//
#include <hip/hip_runtime.h>

// ---------------------------------------------------------------------------
// CDNA5 (gfx1250, wave32) causal self-attention: B=4, T=2048, C=1024, H=16, D=64
// Pipeline: cast(f32->bf16) -> GEMM(qkv) -> flash-attention -> GEMM(proj)
// Matmuls: v_wmma_f32_16x16x32_bf16. GEMM staging: global_load_async_to_lds_b128
// (ASYNCcnt) with double-buffered LDS, one barrier per K-step.
// ---------------------------------------------------------------------------

typedef __bf16 bf16_t;
typedef bf16_t v16bf __attribute__((ext_vector_type(16)));
typedef bf16_t v8bf  __attribute__((ext_vector_type(8)));
typedef float  v8f   __attribute__((ext_vector_type(8)));

#define WMMA_BF16(a, b, c) \
    __builtin_amdgcn_wmma_f32_16x16x32_bf16(false, (a), false, (b), (short)0, (c), false, false)

__device__ __forceinline__ bf16_t f2bf(float f) {
    unsigned u = __float_as_uint(f);
    unsigned r = u + 0x7FFFu + ((u >> 16) & 1u);   // round-to-nearest-even
    unsigned short hv = (unsigned short)(r >> 16);
    return __builtin_bit_cast(bf16_t, hv);
}

// A-operand fragment (16x32 bf16, MxK): lane holds row (caller adds row*ld),
// elements e=0..7 -> K=8h+e ; e=8..15 -> K=16+8h+(e-8).   (ISA 7.12.2)
__device__ __forceinline__ v16bf frag_a(const bf16_t* p, int h) {
    v8bf lo = *(const v8bf*)(p + 8 * h);
    v8bf hi = *(const v8bf*)(p + 16 + 8 * h);
    return __builtin_shufflevector(lo, hi, 0,1,2,3,4,5,6,7,8,9,10,11,12,13,14,15);
}

// B-operand fragment (32x16 bf16, KxN), source stored row-major as [N][K]:
// lane holds col n=L&15; e=0..7 -> K=8h+e ; e=8..15 -> K=16+8h+(e-8).
__device__ __forceinline__ v16bf frag_b(const bf16_t* p, int h) {
    v8bf lo = *(const v8bf*)(p + 8 * h);
    v8bf hi = *(const v8bf*)(p + 16 + 8 * h);
    return __builtin_shufflevector(lo, hi, 0,1,2,3,4,5,6,7,8,9,10,11,12,13,14,15);
}

// Async DMA one 64-byte row (32 bf16) global -> LDS. INST_OFFSET applies to
// both LDS and global addresses (ISA 08 §4.4), so one base covers the row.
__device__ __forceinline__ void async_row64(unsigned ldsa, const bf16_t* g) {
    asm volatile(
        "global_load_async_to_lds_b128 %0, %1, off\n\t"
        "global_load_async_to_lds_b128 %0, %1, off offset:16\n\t"
        "global_load_async_to_lds_b128 %0, %1, off offset:32\n\t"
        "global_load_async_to_lds_b128 %0, %1, off offset:48"
        :: "v"(ldsa), "v"(g) : "memory");
}
__device__ __forceinline__ void wait_async0() {
    asm volatile("s_wait_asynccnt 0x0" ::: "memory");
}
__device__ __forceinline__ unsigned lds_addr(const void* p) {
    return (unsigned)(uintptr_t)p;     // generic LDS ptr: addr[31:0] = LDS offset
}

// ---------------------------------------------------------------------------
__global__ void cast_f32_bf16(const float* __restrict__ src, bf16_t* __restrict__ dst, int n) {
    int i = blockIdx.x * blockDim.x + threadIdx.x;
    int stride = gridDim.x * blockDim.x;
    for (; i < n; i += stride) dst[i] = f2bf(src[i]);
}

// ---------------------------------------------------------------------------
// C[m,n] = sum_k A[m,k] * Bw[n,k].  A:[M][K] bf16, Bw:[N][K] bf16.
// Block = 128 threads (4 waves, 2x2). Block tile 128Mx128N, wave tile 64x64.
// K-step 32, async double-buffered LDS, 16 WMMA per wave per K-step.
// ---------------------------------------------------------------------------
__global__ __launch_bounds__(128) void gemm_bf16(const bf16_t* __restrict__ A,
                                                 const bf16_t* __restrict__ Bw,
                                                 float* __restrict__ Cf,
                                                 bf16_t* __restrict__ Cb,
                                                 int N, int K) {
    __shared__ __align__(16) bf16_t As[2][128 * 40];   // 32 cols + 8 pad
    __shared__ __align__(16) bf16_t Bs[2][128 * 40];

    const int tid  = threadIdx.x;
    const int lane = tid & 31;
    const int w    = tid >> 5;
    const int h    = lane >> 4;
    const int ln   = lane & 15;
    const int wm   = (w >> 1) * 64;    // wave M offset in block tile
    const int wn   = (w & 1) * 64;     // wave N offset in block tile
    const long m0  = (long)blockIdx.y * 128;
    const long n0  = (long)blockIdx.x * 128;

    v8f acc[4][4] = {};

    // staging: thread t owns row t (64B) of both A and B tiles
    const bf16_t* gA = A  + (m0 + tid) * K;
    const bf16_t* gB = Bw + (n0 + tid) * K;
    const unsigned la = lds_addr(&As[0][tid * 40]);
    const unsigned lb = lds_addr(&Bs[0][tid * 40]);
    const unsigned bufstep = 128 * 40 * 2;             // bytes between buffers

    // prologue: fill buffer 0
    async_row64(la, gA);
    async_row64(lb, gB);

    int p = 0;
    for (int k0 = 0; k0 < K; k0 += 32) {
        wait_async0();
        __syncthreads();
        if (k0 + 32 < K) {                             // prefetch next K-slab
            async_row64(la + (p ^ 1) * bufstep, gA + k0 + 32);
            async_row64(lb + (p ^ 1) * bufstep, gB + k0 + 32);
        }
        v16bf af[4], bfr[4];
#pragma unroll
        for (int i = 0; i < 4; ++i)
            af[i] = frag_a(&As[p][(wm + i * 16 + ln) * 40], h);
#pragma unroll
        for (int j = 0; j < 4; ++j)
            bfr[j] = frag_b(&Bs[p][(wn + j * 16 + ln) * 40], h);
#pragma unroll
        for (int i = 0; i < 4; ++i)
#pragma unroll
            for (int j = 0; j < 4; ++j)
                acc[i][j] = WMMA_BF16(af[i], bfr[j], acc[i][j]);
        p ^= 1;
    }

#pragma unroll
    for (int i = 0; i < 4; ++i)
#pragma unroll
        for (int j = 0; j < 4; ++j)
#pragma unroll
            for (int r = 0; r < 8; ++r) {
                long m = m0 + wm + i * 16 + r + 8 * h;
                long n = n0 + wn + j * 16 + ln;
                float v = acc[i][j][r];
                if (Cf) Cf[m * N + n] = v;
                else    Cb[m * N + n] = f2bf(v);
            }
}

// ---------------------------------------------------------------------------
// Flash attention. qkv:[B*T][3C] bf16 (Q|K|V per row). ao:[B*T][C] bf16.
// grid = (T/64, H, B), block = 128 (4 waves); wave w owns q rows q0..q0+15.
// ---------------------------------------------------------------------------
__global__ __launch_bounds__(128) void attn_fwd(const bf16_t* __restrict__ qkv,
                                                bf16_t* __restrict__ ao) {
    constexpr int Tn = 2048, Cc = 1024, HD = 64, C3 = 3072;
    __shared__ __align__(16) bf16_t Ks[32 * 72];      // [key][d], 64 cols + 8 pad
    __shared__ __align__(16) bf16_t Vt[64 * 40];      // [d][key], 32 cols + 8 pad
    __shared__ __align__(16) bf16_t Ps[4][16 * 40];   // per-wave P tile [q][key]

    const int tid  = threadIdx.x;
    const int lane = tid & 31;
    const int w    = tid >> 5;
    const int h    = lane >> 4;
    const int ln   = lane & 15;
    const int b    = blockIdx.z;
    const int head = blockIdx.y;
    const int q0   = blockIdx.x * 64 + w * 16;
    const long rowbase = (long)b * Tn;

    const bf16_t* qp = qkv + (rowbase + q0 + ln) * C3 + head * HD;
    v16bf qf0 = frag_a(qp, h);
    v16bf qf1 = frag_a(qp + 32, h);

    v8f acc[4] = {};
    float mrun[8], lrun[8];
#pragma unroll
    for (int r = 0; r < 8; ++r) { mrun[r] = -1e30f; lrun[r] = 0.0f; }

    const float sc_log2 = 0.125f * 1.44269504088896340736f;  // 1/sqrt(64)*log2(e)
    const int kend = blockIdx.x * 64 + 63;
    const int kr = tid >> 2;            // 0..31 key row for staging
    const int dc = (tid & 3) * 16;      // dim chunk for staging

    for (int j0 = 0; j0 <= kend; j0 += 32) {
        const bf16_t* kg = qkv + (rowbase + j0 + kr) * C3 + Cc + head * HD + dc;
        const bf16_t* vg = qkv + (rowbase + j0 + kr) * C3 + 2 * Cc + head * HD + dc;
        v8bf k0v = *(const v8bf*)kg;  v8bf k1v = *(const v8bf*)(kg + 8);
        v8bf v0v = *(const v8bf*)vg;  v8bf v1v = *(const v8bf*)(vg + 8);
        __syncthreads();
        *(v8bf*)(Ks + kr * 72 + dc)     = k0v;
        *(v8bf*)(Ks + kr * 72 + dc + 8) = k1v;
#pragma unroll
        for (int e = 0; e < 8; ++e) {
            Vt[(dc + e) * 40 + kr]     = v0v[e];
            Vt[(dc + 8 + e) * 40 + kr] = v1v[e];
        }
        __syncthreads();

        if (j0 <= q0 + 15) {            // wave-uniform: EXEC stays full
            v8f s[2] = {};
#pragma unroll
            for (int nkt = 0; nkt < 2; ++nkt) {
                const bf16_t* kp = Ks + (nkt * 16 + ln) * 72;
                s[nkt] = WMMA_BF16(qf0, frag_b(kp, h), s[nkt]);
                s[nkt] = WMMA_BF16(qf1, frag_b(kp + 32, h), s[nkt]);
            }
            bf16_t* pw = &Ps[w][0];
#pragma unroll
            for (int r = 0; r < 8; ++r) {
                int m_g = q0 + r + 8 * h;
                float s0 = (j0 + ln      <= m_g) ? s[0][r] * sc_log2 : -1e30f;
                float s1 = (j0 + 16 + ln <= m_g) ? s[1][r] * sc_log2 : -1e30f;
                float v = fmaxf(s0, s1);
                v = fmaxf(v, __shfl_xor(v, 1, 32));
                v = fmaxf(v, __shfl_xor(v, 2, 32));
                v = fmaxf(v, __shfl_xor(v, 4, 32));
                v = fmaxf(v, __shfl_xor(v, 8, 32));
                float mn    = fmaxf(mrun[r], v);
                float scale = exp2f(mrun[r] - mn);
                float p0 = exp2f(s0 - mn);
                float p1 = exp2f(s1 - mn);
                float rs = p0 + p1;
                rs += __shfl_xor(rs, 1, 32);
                rs += __shfl_xor(rs, 2, 32);
                rs += __shfl_xor(rs, 4, 32);
                rs += __shfl_xor(rs, 8, 32);
                lrun[r] = lrun[r] * scale + rs;
                mrun[r] = mn;
                acc[0][r] *= scale; acc[1][r] *= scale;
                acc[2][r] *= scale; acc[3][r] *= scale;
                pw[(r + 8 * h) * 40 + ln]      = f2bf(p0);
                pw[(r + 8 * h) * 40 + ln + 16] = f2bf(p1);
            }
            v16bf pf = frag_a(pw + ln * 40, h);   // same-wave LDS RAW: in-order
#pragma unroll
            for (int db = 0; db < 4; ++db) {
                v16bf vf = frag_b(Vt + (db * 16 + ln) * 40, h);
                acc[db] = WMMA_BF16(pf, vf, acc[db]);
            }
        }
    }

#pragma unroll
    for (int r = 0; r < 8; ++r) {
        float inv = 1.0f / lrun[r];
        long m = rowbase + q0 + r + 8 * h;
#pragma unroll
        for (int db = 0; db < 4; ++db)
            ao[m * Cc + head * HD + db * 16 + ln] = f2bf(acc[db][r] * inv);
    }
}

// ---------------------------------------------------------------------------
extern "C" void kernel_launch(void* const* d_in, const int* in_sizes, int n_in,
                              void* d_out, int out_size, void* d_ws, size_t ws_size,
                              hipStream_t stream) {
    (void)in_sizes; (void)n_in; (void)out_size; (void)ws_size;
    constexpr int B = 4, T = 2048, C = 1024, H = 16;
    constexpr long M = (long)B * T;          // 8192
    const float* x      = (const float*)d_in[0];
    const float* w_attn = (const float*)d_in[1];
    const float* w_proj = (const float*)d_in[2];
    float* out = (float*)d_out;

    char* ws = (char*)d_ws;
    size_t off = 0;
    auto carve = [&](size_t bytes) { char* p = ws + off; off = (off + bytes + 255) & ~(size_t)255; return p; };
    bf16_t* xb  = (bf16_t*)carve(M * C * 2);
    bf16_t* wab = (bf16_t*)carve((size_t)3 * C * C * 2);
    bf16_t* wpb = (bf16_t*)carve((size_t)C * C * 2);
    bf16_t* qkv = (bf16_t*)carve(M * 3 * C * 2);
    bf16_t* ao  = (bf16_t*)carve(M * C * 2);

    cast_f32_bf16<<<2048, 256, 0, stream>>>(x,      xb,  (int)(M * C));
    cast_f32_bf16<<<2048, 256, 0, stream>>>(w_attn, wab, 3 * C * C);
    cast_f32_bf16<<<2048, 256, 0, stream>>>(w_proj, wpb, C * C);

    // qkv = x @ w_attn^T : M=8192, N=3072, K=1024
    gemm_bf16<<<dim3(3 * C / 128, M / 128), 128, 0, stream>>>(xb, wab, nullptr, qkv, 3 * C, C);

    // flash attention
    attn_fwd<<<dim3(T / 64, H, B), 128, 0, stream>>>(qkv, ao);

    // out = ao @ w_proj^T : M=8192, N=1024, K=1024 (fp32 out)
    gemm_bf16<<<dim3(C / 128, M / 128), 128, 0, stream>>>(ao, wpb, out, nullptr, C, C);
}
